// PointnetFPModule_876173328640
// MI455X (gfx1250) — matrix operand order repaired and checked by reading, not compiled, not run on hardware
//
#include <hip/hip_runtime.h>
#include <hip/hip_bf16.h>

// ---------------- problem constants ----------------
#define Bk   8
#define Nk   8192
#define Mk   1024
#define C1k  256
#define C2k  512
#define CINk 768
#define Hk   512
#define BN_EPS 1e-5f

typedef __bf16 v16bf __attribute__((ext_vector_type(16)));
typedef float  v8f   __attribute__((ext_vector_type(8)));
typedef int    v4i   __attribute__((vector_size(16)));

#ifndef __has_builtin
#define __has_builtin(x) 0
#endif
#if __has_builtin(__builtin_amdgcn_global_load_async_to_lds_b128) && \
    __has_builtin(__builtin_amdgcn_s_wait_asynccnt)
#define HAVE_ASYNC_LDS 1
#define AS1 __attribute__((address_space(1)))
#define AS3 __attribute__((address_space(3)))
#else
#define HAVE_ASYNC_LDS 0
#endif

__device__ __forceinline__ __bf16 f2bf(float f) {
    unsigned u = __builtin_bit_cast(unsigned, f);
    u += 0x7FFFu + ((u >> 16) & 1u);            // round-to-nearest-even
    unsigned short h = (unsigned short)(u >> 16);
    return __builtin_bit_cast(__bf16, h);
}

// ---------------- fp32 -> bf16 weight conversion ----------------
__global__ __launch_bounds__(256) void k_cvt_bf(const float* __restrict__ src,
                                                __bf16* __restrict__ dst, int n) {
    int i = blockIdx.x * 256 + threadIdx.x;
    if (i < n) dst[i] = f2bf(src[i]);
}

// ---------------- three_nn + inverse-distance weights ----------------
__global__ __launch_bounds__(256) void k_three_nn(
    const float* __restrict__ unknown,  // B x N x 3
    const float* __restrict__ known,    // B x M x 3
    int*   __restrict__ idx_out,        // B x N x 3
    float* __restrict__ w_out)          // B x N x 3
{
    __shared__ float kx[Mk], ky[Mk], kz[Mk];
    const int blocksPerB = Nk / 256;
    const int b = blockIdx.x / blocksPerB;
    const int n = (blockIdx.x % blocksPerB) * 256 + threadIdx.x;

    const float* kb = known + (size_t)b * Mk * 3;
    for (int i = threadIdx.x; i < Mk; i += 256) {
        kx[i] = kb[i * 3 + 0];
        ky[i] = kb[i * 3 + 1];
        kz[i] = kb[i * 3 + 2];
    }
    __syncthreads();

    const float* up = unknown + ((size_t)b * Nk + n) * 3;
    const float ux = up[0], uy = up[1], uz = up[2];

    float d0 = 3.0e38f, d1 = 3.0e38f, d2 = 3.0e38f;
    int   i0 = 0, i1 = 0, i2 = 0;
    for (int m = 0; m < Mk; ++m) {
        float dx = ux - kx[m], dy = uy - ky[m], dz = uz - kz[m];
        float d = dx * dx + dy * dy + dz * dz;
        if (d < d0)      { d2 = d1; i2 = i1; d1 = d0; i1 = i0; d0 = d; i0 = m; }
        else if (d < d1) { d2 = d1; i2 = i1; d1 = d;  i1 = m; }
        else if (d < d2) { d2 = d;  i2 = m; }
    }
    float r0 = 1.0f / (fmaxf(d0, 0.0f) + 1e-8f);
    float r1 = 1.0f / (fmaxf(d1, 0.0f) + 1e-8f);
    float r2 = 1.0f / (fmaxf(d2, 0.0f) + 1e-8f);
    float inv = 1.0f / (r0 + r1 + r2);

    size_t o = ((size_t)b * Nk + n) * 3;
    idx_out[o + 0] = i0; idx_out[o + 1] = i1; idx_out[o + 2] = i2;
    w_out[o + 0] = r0 * inv; w_out[o + 1] = r1 * inv; w_out[o + 2] = r2 * inv;
}

// ---------------- interpolate + concat -> bf16 feats, layout (B, N, CIN) ----------------
__global__ __launch_bounds__(256) void k_build_feats(
    const float* __restrict__ known_feats,   // B x C2 x M
    const float* __restrict__ unknow_feats,  // B x C1 x N
    const int*   __restrict__ idx,           // B x N x 3
    const float* __restrict__ wgt,           // B x N x 3
    __bf16* __restrict__ feats)              // B x N x CIN  (K-contiguous for GEMM B-operand)
{
    const int gid = blockIdx.x * 256 + threadIdx.x;   // over B*N
    const int b = gid / Nk, n = gid % Nk;
    const size_t g3 = (size_t)gid * 3;
    const int i0 = idx[g3 + 0], i1 = idx[g3 + 1], i2 = idx[g3 + 2];
    const float w0 = wgt[g3 + 0], w1 = wgt[g3 + 1], w2 = wgt[g3 + 2];

    const float* kf = known_feats + (size_t)b * C2k * Mk;
    __bf16* fb = feats + (size_t)gid * CINk;
    #pragma unroll 4
    for (int c = 0; c < C2k; ++c) {
        const float* row = kf + (size_t)c * Mk;
        float v = w0 * row[i0] + w1 * row[i1] + w2 * row[i2];
        fb[c] = f2bf(v);
    }
    const float* uf = unknow_feats + (size_t)b * C1k * Nk;
    #pragma unroll 4
    for (int c = 0; c < C1k; ++c)
        fb[C2k + c] = f2bf(uf[(size_t)c * Nk + n]);
}

// ---------------- bf16 WMMA GEMM: OUT[b][o][n] = sum_c A[o][c] * X[b][n][c] ----------------
// Block tile 64(o) x 64(n), K step 32, double-buffered async-to-LDS staging.
// LDS tiles are stored pre-packed in WMMA per-lane fragment order (ISA 7.12.2):
//   A (16x32, 16-bit): lane = m + 16*((k>>3)&1), elem j = (k&7) + 8*(k>>4)
//   B (32x16, 16-bit): lane = n + 16*((k>>3)&1), elem j = (k&7) + 8*(k>>4)
// so every 16B global chunk lands as a contiguous 16B LDS chunk (1 async b128/lane),
// and each WMMA operand is a single aligned 32B LDS read (2x ds_load_b128).
#define BM  64
#define BNt 64
#define BKt 32

__global__ __launch_bounds__(256) void k_gemm_bf16(
    const __bf16* __restrict__ A,    // Hd x Kd row-major (weights)
    const __bf16* __restrict__ X,    // B x Nk x Kd (activations, K-contiguous)
    float* __restrict__ OUT,         // B x Hd x Nk
    int Hd, int Kd)
{
    __shared__ __align__(32) __bf16 Apk[2][4 * 32 * 16];   // [buf][tile(4)][lane(32)][elem(16)]
    __shared__ __align__(32) __bf16 Bpk[2][4 * 32 * 16];

    const int b  = blockIdx.z;
    const int m0 = blockIdx.y * BM;
    const int n0 = blockIdx.x * BNt;

    const int tid  = threadIdx.x;
    const int wave = tid >> 5;
    const int lane = tid & 31;

    const int tm = (wave & 3) * 16;        // 16-row band within block tile
    const int tn = (wave >> 2) * 32;       // 0 or 32: two 16-col tiles

    const __bf16* Xb = X + (size_t)b * Nk * Kd;

    // --- per-thread staging chunk (one 16B A chunk + one 16B B chunk per stage) ---
    const int rA  = tid >> 2;              // A row (o) within block tile, 0..63
    const int kqA = tid & 3;               // which 8-element k quad
    const int eA  = (((rA >> 4) * 32) + ((rA & 15) + 16 * (kqA & 1))) * 16 + 8 * (kqA >> 1);
    const __bf16* Agbase = A + (size_t)(m0 + rA) * Kd + 8 * kqA;

    const int nB  = tid >> 2;              // B column (n) within block tile, 0..63
    const int kqB = tid & 3;
    const int eB  = (((nB >> 4) * 32) + ((nB & 15) + 16 * (kqB & 1))) * 16 + 8 * (kqB >> 1);
    const __bf16* Bgbase = Xb + (size_t)(n0 + nB) * Kd + 8 * kqB;

    const int nStages = Kd / BKt;

    auto stage = [&](int s, int buf) {
        const __bf16* Ag = Agbase + s * BKt;
        const __bf16* Bg = Bgbase + s * BKt;
        __bf16* Al = &Apk[buf][eA];
        __bf16* Bl = &Bpk[buf][eB];
#if HAVE_ASYNC_LDS
        __builtin_amdgcn_global_load_async_to_lds_b128((AS1 v4i*)Ag, (AS3 v4i*)Al, 0, 0);
        __builtin_amdgcn_global_load_async_to_lds_b128((AS1 v4i*)Bg, (AS3 v4i*)Bl, 0, 0);
#else
        uint4 av = *(const uint4*)Ag;
        uint4 bv = *(const uint4*)Bg;
        *(uint4*)Al = av;
        *(uint4*)Bl = bv;
#endif
    };

    v8f acc0 = {}, acc1 = {};

    stage(0, 0);
    for (int s = 0; s < nStages; ++s) {
        const int buf = s & 1;
        if (s + 1 < nStages) {
            stage(s + 1, buf ^ 1);
#if HAVE_ASYNC_LDS
            __builtin_amdgcn_s_wait_asynccnt(2);   // stage s landed; s+1 still in flight
#endif
        } else {
#if HAVE_ASYNC_LDS
            __builtin_amdgcn_s_wait_asynccnt(0);
#endif
        }
        __syncthreads();

        v16bf af   = *(const v16bf*)&Apk[buf][(((tm >> 4)) * 32 + lane) * 16];
        v16bf bfr0 = *(const v16bf*)&Bpk[buf][(((tn >> 4)) * 32 + lane) * 16];
        v16bf bfr1 = *(const v16bf*)&Bpk[buf][(((tn >> 4) + 1) * 32 + lane) * 16];

        acc0 = __builtin_amdgcn_wmma_f32_16x16x32_bf16(false, af, false, bfr0,
                                                       (short)0, acc0, false, false);
        acc1 = __builtin_amdgcn_wmma_f32_16x16x32_bf16(false, af, false, bfr1,
                                                       (short)0, acc1, false, false);
        __syncthreads();
    }

    // D layout (32-bit C/D 16x16): lanes 0-15 -> M = v, lanes 16-31 -> M = 8+v; N = lane&15
    float* Ob = OUT + (size_t)b * Hd * Nk;
    const int Mbase = (lane >> 4) * 8;
    const int Ncol  = lane & 15;
    #pragma unroll
    for (int v = 0; v < 8; ++v) {
        int Mrow = Mbase + v;
        Ob[(size_t)(m0 + tm + Mrow) * Nk + (n0 + tn + Ncol)]      = acc0[v];
        Ob[(size_t)(m0 + tm + Mrow) * Nk + (n0 + tn + 16 + Ncol)] = acc1[v];
    }
}

// ---------------- BN statistics: one block per channel ----------------
__global__ __launch_bounds__(256) void k_bn_stats(
    const float* __restrict__ X,   // B x Hd x N
    float* __restrict__ mean, float* __restrict__ rstd, int Hd)
{
    const int c = blockIdx.x;
    float s = 0.0f, s2 = 0.0f;
    for (int b = 0; b < Bk; ++b) {
        const float* p = X + ((size_t)b * Hd + c) * Nk;
        for (int n = threadIdx.x; n < Nk; n += 256) {
            float v = p[n];
            s += v; s2 += v * v;
        }
    }
    __shared__ float sh[256], sh2[256];
    sh[threadIdx.x] = s; sh2[threadIdx.x] = s2;
    __syncthreads();
    for (int stride = 128; stride > 0; stride >>= 1) {
        if (threadIdx.x < stride) {
            sh[threadIdx.x]  += sh[threadIdx.x + stride];
            sh2[threadIdx.x] += sh2[threadIdx.x + stride];
        }
        __syncthreads();
    }
    if (threadIdx.x == 0) {
        const float cnt = (float)(Bk * Nk);
        float m   = sh[0] / cnt;
        float var = sh2[0] / cnt - m * m;
        mean[c] = m;
        rstd[c] = rsqrtf(var + BN_EPS);
    }
}

// ---- normalize + affine + ReLU, transpose (B,H,N) -> (B,N,H), cast bf16 ----
__global__ __launch_bounds__(256) void k_bn_relu_bf_t(
    const float* __restrict__ X, const float* __restrict__ mean,
    const float* __restrict__ rstd, const float* __restrict__ g,
    const float* __restrict__ beta, __bf16* __restrict__ Y, int Hd)
{
    size_t i = (size_t)blockIdx.x * 256 + threadIdx.x;   // over B*Hd*N, n fastest (coalesced read)
    int n = (int)(i % Nk);
    size_t t = i / Nk;
    int c = (int)(t % Hd);
    int b = (int)(t / Hd);
    float v = (X[i] - mean[c]) * rstd[c] * g[c] + beta[c];
    Y[((size_t)b * Nk + n) * Hd + c] = f2bf(fmaxf(v, 0.0f));
}

// ---- normalize + affine + ReLU in place (final fp32 output, (B,H,N)) ----
__global__ __launch_bounds__(256) void k_bn_relu_f32(
    float* __restrict__ X, const float* __restrict__ mean,
    const float* __restrict__ rstd, const float* __restrict__ g,
    const float* __restrict__ beta, int Hd)
{
    size_t i = (size_t)blockIdx.x * 256 + threadIdx.x;
    int c = (int)((i / Nk) % Hd);
    float v = (X[i] - mean[c]) * rstd[c] * g[c] + beta[c];
    X[i] = fmaxf(v, 0.0f);
}

// ---------------- host launcher ----------------
extern "C" void kernel_launch(void* const* d_in, const int* in_sizes, int n_in,
                              void* d_out, int out_size, void* d_ws, size_t ws_size,
                              hipStream_t stream) {
    const float* unknown      = (const float*)d_in[0];
    const float* known        = (const float*)d_in[1];
    const float* unknow_feats = (const float*)d_in[2];
    const float* known_feats  = (const float*)d_in[3];
    const float* W1 = (const float*)d_in[4];
    const float* g1 = (const float*)d_in[5];
    const float* b1 = (const float*)d_in[6];
    const float* W2 = (const float*)d_in[7];
    const float* g2 = (const float*)d_in[8];
    const float* b2 = (const float*)d_in[9];
    float* out = (float*)d_out;

    // ---- workspace layout (256B-aligned regions) ----
    char* ws = (char*)d_ws;
    size_t off = 0;
    auto carve = [&](size_t bytes) { void* p = ws + off; off += (bytes + 255) & ~(size_t)255; return p; };

    int*    idx    = (int*)   carve((size_t)Bk * Nk * 3 * sizeof(int));
    float*  wgt    = (float*) carve((size_t)Bk * Nk * 3 * sizeof(float));
    __bf16* W1bf   = (__bf16*)carve((size_t)Hk * CINk * sizeof(__bf16));
    __bf16* W2bf   = (__bf16*)carve((size_t)Hk * Hk   * sizeof(__bf16));
    float*  mean1  = (float*) carve(Hk * sizeof(float));
    float*  rstd1  = (float*) carve(Hk * sizeof(float));
    float*  mean2  = (float*) carve(Hk * sizeof(float));
    float*  rstd2  = (float*) carve(Hk * sizeof(float));
    __bf16* feats  = (__bf16*)carve((size_t)Bk * Nk * CINk * sizeof(__bf16)); // 100.7 MB
    float*  h_pre  = (float*) carve((size_t)Bk * Hk * Nk * sizeof(float));    // 134 MB
    __bf16* h_bf   = feats;  // reuse: feats dead after GEMM1 (67 MB < 100.7 MB)

    // 1) weights -> bf16
    k_cvt_bf<<<(Hk * CINk + 255) / 256, 256, 0, stream>>>(W1, W1bf, Hk * CINk);
    k_cvt_bf<<<(Hk * Hk + 255) / 256, 256, 0, stream>>>(W2, W2bf, Hk * Hk);

    // 2) three_nn + weights
    k_three_nn<<<Bk * (Nk / 256), 256, 0, stream>>>(unknown, known, idx, wgt);

    // 3) interpolate + concat -> bf16 feats (B, N, CIN)
    k_build_feats<<<(Bk * Nk) / 256, 256, 0, stream>>>(known_feats, unknow_feats, idx, wgt, feats);

    // 4) GEMM1 (WMMA bf16): h_pre = W1 * feats^T
    {
        dim3 grid(Nk / BNt, Hk / BM, Bk);
        k_gemm_bf16<<<grid, 256, 0, stream>>>(W1bf, feats, h_pre, Hk, CINk);
    }

    // 5) BN1 stats, then normalize+ReLU -> bf16 h (B, N, H)
    k_bn_stats<<<Hk, 256, 0, stream>>>(h_pre, mean1, rstd1, Hk);
    k_bn_relu_bf_t<<<(unsigned)(((size_t)Bk * Hk * Nk) / 256), 256, 0, stream>>>(
        h_pre, mean1, rstd1, g1, b1, h_bf, Hk);

    // 6) GEMM2 (WMMA bf16): out(pre-act) = W2 * h^T
    {
        dim3 grid(Nk / BNt, Hk / BM, Bk);
        k_gemm_bf16<<<grid, 256, 0, stream>>>(W2bf, h_bf, out, Hk, Hk);
    }

    // 7) BN2 stats, then normalize+ReLU in place on d_out
    k_bn_stats<<<Hk, 256, 0, stream>>>(out, mean2, rstd2, Hk);
    k_bn_relu_f32<<<(unsigned)(((size_t)Bk * Hk * Nk) / 256), 256, 0, stream>>>(
        out, mean2, rstd2, g2, b2, Hk);

    (void)in_sizes; (void)n_in; (void)out_size; (void)ws_size;
}
